// LinformerSelfAttention_71975061946524
// MI455X (gfx1250) — matrix operand order repaired and checked by reading, not compiled, not run on hardware
//
#include <hip/hip_runtime.h>

// ---------------------------------------------------------------------------
// Linformer self-attention (bug-faithful reference) for MI455X / gfx1250.
//
// Algebraic simplification: the reference einsum contracts over n, so
// k_proj[b,h,k,d] = E[k,d]*ksum[b,h,d] with ksum = (sum_n x[b,n,:]) @ Wk.T
// (same for v_proj). The K and V 16384x1024x1024 GEMMs are skipped; ksum*(1/8)
// folds into the Q epilogue, vsum into the AV epilogue.
//
// Matrix stages: v_wmma_f32_16x16x32_bf16 (wave32 layouts, ISA 7.12.2).
// Big GEMMs stage A/B panels into LDS with the Tensor Data Mover
// (tensor_load_to_lds + s_wait_tensorcnt), double-buffered.
// ---------------------------------------------------------------------------

typedef unsigned short u16;
typedef __bf16 bf16;
typedef bf16  v16bf __attribute__((ext_vector_type(16)));
typedef float v8f   __attribute__((ext_vector_type(8)));
typedef unsigned int tdm_v4u __attribute__((ext_vector_type(4)));
typedef int          tdm_v8i __attribute__((ext_vector_type(8)));
typedef int          tdm_v4i __attribute__((ext_vector_type(4)));

#define DIM  1024
#define NSEQ 4096
#define NB   4
#define NH   16
#define HD   64
#define KLIN 256

__device__ __forceinline__ u16 f32_to_bf16_rne(float f) {
  unsigned u = __float_as_uint(f);
  u += 0x7FFFu + ((u >> 16) & 1u);      // round-to-nearest-even
  return (u16)(u >> 16);
}

union V16 { uint4 q[2]; v16bf v; };
union V8F { v8f v; float f[8]; };

__device__ __forceinline__ v16bf load2x16(const u16* p0, const u16* p1) {
  V16 t;
  t.q[0] = *(const uint4*)p0;
  t.q[1] = *(const uint4*)p1;
  return t.v;
}

__device__ __forceinline__ v8f wmma_bf16(v16bf a, v16bf b, v8f c) {
  // (neg_a, A, neg_b, B, c_mod, C, reuse_a, reuse_b)
  return __builtin_amdgcn_wmma_f32_16x16x32_bf16(false, a, false, b,
                                                 (short)0, c, false, false);
}

// ---------------------------------------------------------------------------
// TDM: DMA a 2-D tile (tile_h rows x tile_w bf16 elems, row stride in elems)
// from global memory into LDS at lds_byte_addr. D# per ISA 08_async_tensor §8.
// ---------------------------------------------------------------------------
__device__ __forceinline__ void tdm_load_tile_2d(unsigned lds_byte_addr,
                                                 const u16* gsrc, int tile_w,
                                                 int tile_h,
                                                 int row_stride_elems) {
  unsigned long long ga = (unsigned long long)(uintptr_t)gsrc;
  tdm_v4u g0;
  g0[0] = 1u;                                  // count=1, user descriptor
  g0[1] = lds_byte_addr;                       // lds_addr (bytes)
  g0[2] = (unsigned)ga;                        // global_addr[31:0]
  g0[3] = (unsigned)((ga >> 32) & 0x01FFFFFFu) // global_addr[56:32]
          | (2u << 30);                        // type=2 ("image")
  tdm_v8i g1;
  g1[0] = 1 << 16;                             // data_size=2B; mask=0
  g1[1] = (int)((unsigned)(tile_w & 0xFFFF) << 16);        // tensor_dim0 lo
  g1[2] = (int)(((unsigned)tile_w >> 16) |
                ((unsigned)(tile_h & 0xFFFF) << 16));      // dim0 hi|dim1 lo
  g1[3] = (int)(((unsigned)tile_h >> 16) |
                ((unsigned)tile_w << 16));                 // dim1 hi|tile_dim0
  g1[4] = tile_h;                              // tile_dim1 (tile_dim2=0)
  g1[5] = row_stride_elems;                    // tensor_dim0_stride[31:0]
  g1[6] = 0;
  g1[7] = 0;
  tdm_v4i gz = {};
#if defined(__clang_major__) && __clang_major__ >= 23
  tdm_v8i gz8 = {};
  __builtin_amdgcn_tensor_load_to_lds(g0, g1, gz, gz, gz8, 0);
#else
  __builtin_amdgcn_tensor_load_to_lds(g0, g1, gz, gz, 0);
#endif
}

// ---------------------------------------------------------------------------
// Small prep kernels
// ---------------------------------------------------------------------------
__global__ void cvt_f32_bf16_kernel(const float* __restrict__ in,
                                    u16* __restrict__ out, int n) {
  int i = blockIdx.x * blockDim.x + threadIdx.x;
  if (i < n) out[i] = f32_to_bf16_rne(in[i]);
}

// F (K x HD) -> FT (HD x K) in bf16 so AV's B-matrix columns are contiguous.
__global__ void transpose_f_kernel(const float* __restrict__ F,
                                   u16* __restrict__ FT) {
  int i = blockIdx.x * blockDim.x + threadIdx.x;
  if (i < KLIN * HD) {
    int k = i >> 6, d = i & 63;
    FT[d * KLIN + k] = f32_to_bf16_rne(F[i]);
  }
}

// xsum[b,col] = sum_n x[b,n,col]   (4 x 1024)
__global__ void colsum_x_kernel(const float* __restrict__ x,
                                float* __restrict__ xsum) {
  int i = blockIdx.x * blockDim.x + threadIdx.x;   // [0, 4096)
  int b = i >> 10, col = i & 1023;
  const float* p = x + (size_t)b * NSEQ * DIM + col;
  float s = 0.f;
  for (int n = 0; n < NSEQ; ++n) s += p[(size_t)n * DIM];
  xsum[i] = s;
}

// ksum[b,dd] = 0.125 * sum_i xsum[b,i]*Wk[dd,i] ; vsum likewise (no scale)
__global__ void kv_sums_kernel(const float* __restrict__ xsum,
                               const float* __restrict__ Wk,
                               const float* __restrict__ Wv,
                               float* __restrict__ ksum,
                               float* __restrict__ vsum) {
  int i = blockIdx.x * blockDim.x + threadIdx.x;   // [0, 8192)
  int which = i >> 12;                             // 0 = k, 1 = v
  int bd = i & 4095;
  int b = bd >> 10, dd = bd & 1023;
  const float* W = which ? Wv : Wk;
  const float* xs = xsum + b * DIM;
  const float* w  = W + (size_t)dd * DIM;
  float s = 0.f;
  for (int j = 0; j < DIM; ++j) s += xs[j] * w[j];
  if (which) vsum[bd] = s;
  else       ksum[bd] = s * 0.125f;                // fold 1/sqrt(64)
}

// ---------------------------------------------------------------------------
// GEMM: C(16384 x 1024) = A(bf16 row-major, ld=1024) x W(bf16,(out,in)).T
// Workgroup = 128 threads (4 waves) computing a 64x64 output block.
// K is chunked by 64; the 64x64 A and B panels are DMA'd into LDS by the
// Tensor Data Mover (wave 0 issues; TENSORcnt + barrier hand-off), double
// buffered so the next chunk's DMA overlaps current-chunk WMMAs.
//   MODE 0: out_bf16 = acc * scaleVec[b*1024+col]          (Q path, ksum*1/8)
//   MODE 1: out_f32  = acc + bias[col]                     (output projection)
// ---------------------------------------------------------------------------
template <int MODE>
__global__ __launch_bounds__(128) void gemm16k_bf16_tdm(
    const u16* __restrict__ A, const u16* __restrict__ Bw,
    const float* __restrict__ scaleVec, const float* __restrict__ bias,
    u16* __restrict__ outH, float* __restrict__ outF) {
  __shared__ u16 As[2][64 * 64];                 // 16 KB
  __shared__ u16 Bs[2][64 * 64];                 // 16 KB

  const int lane = threadIdx.x & 31;
  const int wave = threadIdx.x >> 5;
  const int half = lane >> 4;
  const int llo  = lane & 15;
  const int m0 = blockIdx.x * 64;
  const int n0 = blockIdx.y * 64;

  v8f zero = {};
  v8f acc[4] = {zero, zero, zero, zero};

  if (wave == 0) {                               // prologue: chunk 0
    tdm_load_tile_2d((unsigned)(uintptr_t)&As[0][0], A + (size_t)m0 * DIM,
                     64, 64, DIM);
    tdm_load_tile_2d((unsigned)(uintptr_t)&Bs[0][0], Bw + (size_t)n0 * DIM,
                     64, 64, DIM);
    __builtin_amdgcn_s_wait_tensorcnt(0);
  }
  __syncthreads();

  for (int c = 0; c < 16; ++c) {                 // 16 chunks of K=64
    const int cur = c & 1;
    if (wave == 0 && c + 1 < 16) {               // kick next chunk's DMA
      const int k0 = (c + 1) * 64;
      tdm_load_tile_2d((unsigned)(uintptr_t)&As[cur ^ 1][0],
                       A + (size_t)m0 * DIM + k0, 64, 64, DIM);
      tdm_load_tile_2d((unsigned)(uintptr_t)&Bs[cur ^ 1][0],
                       Bw + (size_t)n0 * DIM + k0, 64, 64, DIM);
    }
    const u16* as = &As[cur][0];
    const u16* bs = &Bs[cur][0];
#pragma unroll
    for (int ks = 0; ks < 64; ks += 32) {
      // A 16x32: lane half selects K {0-7,16-23} vs {8-15,24-31}
      const u16* ap = as + (wave * 16 + llo) * 64 + ks + 8 * half;
      v16bf a = load2x16(ap, ap + 16);
#pragma unroll
      for (int nt = 0; nt < 4; ++nt) {
        // B 32x16: lane = column; half selects K {0-7,16-23} vs {8-15,24-31}
        const u16* bp = bs + (nt * 16 + llo) * 64 + ks + 8 * half;
        v16bf b = load2x16(bp, bp + 16);
        acc[nt] = wmma_bf16(a, b, acc[nt]);
      }
    }
    if (wave == 0 && c + 1 < 16) __builtin_amdgcn_s_wait_tensorcnt(0);
    __syncthreads();                             // next buffer ready for all
  }

  const int mw = m0 + wave * 16;
#pragma unroll
  for (int nt = 0; nt < 4; ++nt) {
    V8F c; c.v = acc[nt];
#pragma unroll
    for (int r = 0; r < 8; ++r) {
      int row = mw + r + 8 * half;               // C layout: M = r + 8*half
      int col = n0 + nt * 16 + llo;
      if (MODE == 0) {
        float s = scaleVec[(row >> 12) * DIM + col];   // row>>12 == batch b
        outH[(size_t)row * DIM + col] = f32_to_bf16_rne(c.f[r] * s);
      } else {
        outF[(size_t)row * DIM + col] = c.f[r] + bias[col];
      }
    }
  }
}

// ---------------------------------------------------------------------------
// Fused attention: attn = qk @ E.T (WMMA), softmax over 256,
// then out = attn @ F (WMMA via LDS re-layout), scaled by vsum.
// One wave owns a 16-row tile of one (b,h). 32 + 32 WMMAs per wave.
// ---------------------------------------------------------------------------
__global__ __launch_bounds__(128) void attn_kernel(
    const u16* __restrict__ qk, const u16* __restrict__ Ebf,
    const u16* __restrict__ FT, const float* __restrict__ vsum,
    u16* __restrict__ attnv) {
  __shared__ u16 smem[4 * 16 * KLIN];            // 32 KB, 8 KB per wave

  const int lane = threadIdx.x & 31;
  const int wave = threadIdx.x >> 5;
  const int half = lane >> 4;
  const int llo  = lane & 15;
  const int b = blockIdx.z, h = blockIdx.y;
  const int n0 = blockIdx.x * 64 + wave * 16;

  // ---- phase 1: attn[16 x 256] = qk_tile(16x64) @ E.T ----
  const u16* Qrow = qk + ((size_t)(b * NSEQ + n0 + llo)) * DIM + h * HD;
  v16bf a0 = load2x16(Qrow + 0  + 8 * half, Qrow + 16 + 8 * half);  // d 0..31
  v16bf a1 = load2x16(Qrow + 32 + 8 * half, Qrow + 48 + 8 * half);  // d 32..63

  v8f zero = {};
  V8F acc[16];
#pragma unroll
  for (int t = 0; t < 16; ++t) acc[t].v = zero;

#pragma unroll
  for (int t = 0; t < 16; ++t) {
    // B column kk = E row kk (contiguous over d)
    const u16* e0 = Ebf + (size_t)(t * 16 + llo) * HD + 8 * half;
    v16bf b0 = load2x16(e0, e0 + 16);
    v16bf b1 = load2x16(e0 + 32, e0 + 48);
    acc[t].v = wmma_bf16(a0, b0, acc[t].v);
    acc[t].v = wmma_bf16(a1, b1, acc[t].v);
  }

  // ---- softmax over k=256: reduce over 16 tiles, then 16 lanes of a half ---
  float mx[8], inv[8];
#pragma unroll
  for (int r = 0; r < 8; ++r) {
    float m = acc[0].f[r];
#pragma unroll
    for (int t = 1; t < 16; ++t) m = fmaxf(m, acc[t].f[r]);
#pragma unroll
    for (int s = 1; s < 16; s <<= 1) m = fmaxf(m, __shfl_xor(m, s, 32));
    mx[r] = m;
  }
#pragma unroll
  for (int r = 0; r < 8; ++r) {
    float s = 0.f;
#pragma unroll
    for (int t = 0; t < 16; ++t) {
      float e = __expf(acc[t].f[r] - mx[r]);
      acc[t].f[r] = e;
      s += e;
    }
#pragma unroll
    for (int ss = 1; ss < 16; ss <<= 1) s += __shfl_xor(s, ss, 32);
    inv[r] = 1.f / s;
  }

  // ---- stage normalized attn (bf16) into LDS in row-major A order ----
  u16* sm = smem + wave * (16 * KLIN);
#pragma unroll
  for (int t = 0; t < 16; ++t)
#pragma unroll
    for (int r = 0; r < 8; ++r) {
      int m = r + 8 * half, kk = t * 16 + llo;
      sm[m * KLIN + kk] = f32_to_bf16_rne(acc[t].f[r] * inv[r]);
    }
  // same-wave LDS ops are in-order; no cross-wave sharing -> no barrier

  // ---- phase 2: out(16x64) = attn(16x256) @ F(256x64), scale by vsum ----
#pragma unroll
  for (int dt = 0; dt < 4; ++dt) {
    v8f c = zero;
    for (int k = 0; k < KLIN; k += 32) {
      const u16* ap = sm + llo * KLIN + k + 8 * half;     // ds_load_b128 x2
      v16bf a = load2x16(ap, ap + 16);
      const u16* fp = FT + (size_t)(dt * 16 + llo) * KLIN + k + 8 * half;
      v16bf bb = load2x16(fp, fp + 16);
      c = wmma_bf16(a, bb, c);
    }
    V8F cc; cc.v = c;
#pragma unroll
    for (int r = 0; r < 8; ++r) {
      int m = r + 8 * half;
      int d = dt * 16 + llo;
      float v = cc.f[r] * vsum[b * DIM + h * HD + d];
      attnv[((size_t)(b * NSEQ + n0 + m)) * DIM + h * HD + d] =
          f32_to_bf16_rne(v);
    }
  }
}

// ---------------------------------------------------------------------------
// Host-side orchestration
// ---------------------------------------------------------------------------
extern "C" void kernel_launch(void* const* d_in, const int* in_sizes, int n_in,
                              void* d_out, int out_size, void* d_ws,
                              size_t ws_size, hipStream_t stream) {
  const float* x  = (const float*)d_in[0];   // (4,4096,1024)
  const float* Wq = (const float*)d_in[1];   // (1024,1024)
  const float* Wk = (const float*)d_in[2];
  const float* Wv = (const float*)d_in[3];
  const float* E  = (const float*)d_in[4];   // (256,64)
  const float* F  = (const float*)d_in[5];   // (256,64)
  const float* Wo = (const float*)d_in[6];
  const float* bo = (const float*)d_in[7];   // (1024,)
  float* out = (float*)d_out;

  const size_t NX = (size_t)NB * NSEQ * DIM;        // 16.7M
  char* ws = (char*)d_ws;
  size_t off = 0;
  auto take = [&](size_t bytes) {
    char* p = ws + off;
    off = (off + bytes + 255) & ~(size_t)255;
    return p;
  };
  u16*   x_bf    = (u16*)take(NX * 2);              // 33.5 MB
  u16*   qk_bf   = (u16*)take(NX * 2);              // 33.5 MB
  u16*   av_bf   = (u16*)take(NX * 2);              // 33.5 MB
  u16*   Wq_bf   = (u16*)take((size_t)DIM * DIM * 2);
  u16*   Wo_bf   = (u16*)take((size_t)DIM * DIM * 2);
  u16*   E_bf    = (u16*)take((size_t)KLIN * HD * 2);
  u16*   FT_bf   = (u16*)take((size_t)KLIN * HD * 2);
  float* xsum    = (float*)take((size_t)NB * DIM * 4);
  float* ksum    = (float*)take((size_t)NB * DIM * 4);
  float* vsum    = (float*)take((size_t)NB * DIM * 4);

  // 1) precision conversion
  cvt_f32_bf16_kernel<<<(NX + 255) / 256, 256, 0, stream>>>(x, x_bf, (int)NX);
  cvt_f32_bf16_kernel<<<(DIM * DIM + 255) / 256, 256, 0, stream>>>(
      Wq, Wq_bf, DIM * DIM);
  cvt_f32_bf16_kernel<<<(DIM * DIM + 255) / 256, 256, 0, stream>>>(
      Wo, Wo_bf, DIM * DIM);
  cvt_f32_bf16_kernel<<<(KLIN * HD + 255) / 256, 256, 0, stream>>>(
      E, E_bf, KLIN * HD);
  transpose_f_kernel<<<(KLIN * HD + 255) / 256, 256, 0, stream>>>(F, FT_bf);

  // 2) ksum/vsum via rank-1 structure (skips the K and V GEMMs entirely)
  colsum_x_kernel<<<(NB * DIM) / 256, 256, 0, stream>>>(x, xsum);
  kv_sums_kernel<<<(2 * NB * DIM) / 256, 256, 0, stream>>>(xsum, Wk, Wv,
                                                           ksum, vsum);

  // 3) qk = (x @ Wq.T) * ksum * 1/8   (bf16, WMMA + TDM staging)
  gemm16k_bf16_tdm<0><<<dim3(256, 16), 128, 0, stream>>>(
      x_bf, Wq_bf, ksum, nullptr, qk_bf, nullptr);

  // 4) attn = softmax(qk @ E.T); av = (attn @ F) * vsum   (WMMA x2 + softmax)
  attn_kernel<<<dim3(64, 16, 4), 128, 0, stream>>>(qk_bf, E_bf, FT_bf, vsum,
                                                   av_bf);

  // 5) out = av @ Wo.T + bo   (f32 out, WMMA + TDM staging)
  gemm16k_bf16_tdm<1><<<dim3(256, 16), 128, 0, stream>>>(
      av_bf, Wo_bf, nullptr, bo, nullptr, out);
}
